// RtmScoreHead_51101520888284
// MI455X (gfx1250) — compile-verified, matrix operand order checked by reading, not run on hardware
//
#include <hip/hip_runtime.h>
#include <hip/hip_bf16.h>
#include <math.h>

#define H_DIM 128
#define G_DIM 10
#define NPTS 8192
#define NBATCH 32
#define ROWS_TOTAL (NBATCH * NPTS)        // 262144
#define WAVES_PB 4
#define ROWS_PER_WAVE 32                  // two 16-row A tiles per wave
#define BLOCK_ROWS (WAVES_PB * ROWS_PER_WAVE)   // 128
#define THREADS_PB (WAVES_PB * 32)        // 128

typedef __attribute__((ext_vector_type(16))) __bf16 v16bf;
typedef __attribute__((ext_vector_type(8)))  __bf16 v8bf;
typedef __attribute__((ext_vector_type(8)))  float  v8f;
typedef __attribute__((ext_vector_type(4)))  float  v4f;

// workspace layout (bf16 elements): 3x w1T[128][128], then 3x w2T[16][128] (padded)
#define W1T_ELEMS (H_DIM * H_DIM)                 // 16384
#define W2T_ELEMS (16 * H_DIM)                    // 2048
#define WS_BF16_TOTAL (3 * W1T_ELEMS + 3 * W2T_ELEMS)  // 55296
#define LOSSACC_BYTE_OFF (WS_BF16_TOTAL * 2)      // 110592 (8B aligned)

// ---------------------------------------------------------------------------
// Pre-pass: convert fp32 weights to bf16, transposed into B-matrix layout.
// w1T[n][k] = w1[k][n]; w2T[n][k] = (n<10) ? w2[k][n] : 0
// ---------------------------------------------------------------------------
__global__ void convert_weights(const float* __restrict__ w1_pi,
                                const float* __restrict__ w1_sig,
                                const float* __restrict__ w1_mu,
                                const float* __restrict__ w2_pi,
                                const float* __restrict__ w2_sig,
                                const float* __restrict__ w2_mu,
                                __bf16* __restrict__ ws) {
  int idx = blockIdx.x * blockDim.x + threadIdx.x;
  if (idx >= WS_BF16_TOTAL) return;
  const float* w1s[3] = {w1_pi, w1_sig, w1_mu};
  const float* w2s[3] = {w2_pi, w2_sig, w2_mu};
  if (idx < 3 * W1T_ELEMS) {
    int h = idx / W1T_ELEMS;
    int r = idx % W1T_ELEMS;
    int n = r / H_DIM, k = r % H_DIM;
    ws[idx] = (__bf16)w1s[h][k * H_DIM + n];
  } else {
    int j = idx - 3 * W1T_ELEMS;
    int h = j / W2T_ELEMS;
    int r = j % W2T_ELEMS;
    int n = r / H_DIM, k = r % H_DIM;
    ws[idx] = (n < G_DIM) ? (__bf16)w2s[h][k * G_DIM + n] : (__bf16)0.0f;
  }
}

__global__ void init_acc(float* __restrict__ out_score, float* __restrict__ lossacc) {
  int i = threadIdx.x;
  if (i < NBATCH) out_score[i] = 0.0f;
  if (i < 2) lossacc[i] = 0.0f;
}

__global__ void finalize_loss(const float* __restrict__ lossacc, float* __restrict__ out_loss) {
  out_loss[0] = lossacc[0] / fmaxf(lossacc[1], 1.0f);
}

// B fragment: lane owns column n; khalf splits K per the 16-bit WMMA layout.
__device__ __forceinline__ v16bf load_bfrag(const __bf16* __restrict__ wT, int n, int kbase) {
  const __bf16* p = wT + n * H_DIM + kbase;
  v8bf lo = *(const v8bf*)p;
  v8bf hi = *(const v8bf*)(p + 16);
  return __builtin_shufflevector(lo, hi, 0,1,2,3,4,5,6,7,8,9,10,11,12,13,14,15);
}

// ---------------------------------------------------------------------------
// Fused MDN head: 3x (relu(x@w1+b1)@w2+b2) via bf16 WMMA + per-row epilogue.
// Each wave owns 32 rows (two 16-row A tiles) so every B fragment feeds two
// WMMAs; epilogue uses all 32 lanes (one row per lane).
// ---------------------------------------------------------------------------
__global__ __launch_bounds__(THREADS_PB) void mdn_fused(
    const float* __restrict__ features, const float* __restrict__ dist,
    const unsigned char* __restrict__ dist_mask,
    const float* __restrict__ b1_pi, const float* __restrict__ b2_pi,
    const float* __restrict__ b1_sig, const float* __restrict__ b2_sig,
    const float* __restrict__ b1_mu, const float* __restrict__ b2_mu,
    const __bf16* __restrict__ ws,
    float* __restrict__ out_score, float* __restrict__ out_pi,
    float* __restrict__ out_mu, float* __restrict__ out_sig,
    float* __restrict__ lossacc) {

  __shared__ __bf16 s_hid[WAVES_PB][ROWS_PER_WAVE][H_DIM];   // 32 KB
  __shared__ float  s_epi[WAVES_PB][3][ROWS_PER_WAVE][16];   // 24 KB
  __shared__ float  s_red[WAVES_PB][3];

  const int tid = threadIdx.x;
  const int wave = tid >> 5;
  const int lane = tid & 31;
  const int col = lane & 15;     // column (B/C) or row (A) owned by this lane
  const int khalf = lane >> 4;   // K-half split of the 16-bit fragment layout

  const int rowBase = blockIdx.x * BLOCK_ROWS + wave * ROWS_PER_WAVE;

  // --- Build A fragments for this wave's 32 rows (reused by all 3 heads) ---
  v16bf afrag[2][4];
#pragma unroll
  for (int ra = 0; ra < 2; ++ra) {
    const float* frow = features + (size_t)(rowBase + ra * 16 + col) * H_DIM;
#pragma unroll
    for (int s = 0; s < 4; ++s) {
      const int kb = s * 32 + khalf * 8;
      v4f x0 = *(const v4f*)(frow + kb);
      v4f x1 = *(const v4f*)(frow + kb + 4);
      v4f y0 = *(const v4f*)(frow + kb + 16);
      v4f y1 = *(const v4f*)(frow + kb + 20);
      v16bf a;
#pragma unroll
      for (int j = 0; j < 4; ++j) {
        a[j]      = (__bf16)x0[j];
        a[4 + j]  = (__bf16)x1[j];
        a[8 + j]  = (__bf16)y0[j];
        a[12 + j] = (__bf16)y1[j];
      }
      afrag[ra][s] = a;
    }
  }

  const float* b1s[3] = {b1_pi, b1_sig, b1_mu};
  const float* b2s[3] = {b2_pi, b2_sig, b2_mu};

#pragma unroll 1
  for (int h = 0; h < 3; ++h) {
    const __bf16* w1T = ws + h * W1T_ELEMS;
    const __bf16* w2T = ws + 3 * W1T_ELEMS + h * W2T_ELEMS;

    // Preload this head's b1 biases for all 8 N-tiles (issued as one clause).
    float bias1[8];
#pragma unroll
    for (int nt = 0; nt < 8; ++nt) bias1[nt] = b1s[h][nt * 16 + col];

    // ---- GEMM1: hidden = relu(x @ w1 + b1); each B fragment feeds 2 WMMAs ----
#pragma unroll 1
    for (int nt = 0; nt < 8; ++nt) {
      const int n = nt * 16 + col;
      v8f c0, c1;
#pragma unroll
      for (int r = 0; r < 8; ++r) { c0[r] = bias1[nt]; c1[r] = bias1[nt]; }
#pragma unroll
      for (int s = 0; s < 4; ++s) {
        v16bf b = load_bfrag(w1T, n, s * 32 + khalf * 8);
        c0 = __builtin_amdgcn_wmma_f32_16x16x32_bf16(false, afrag[0][s], false, b,
                                                     (short)0, c0, false, false);
        c1 = __builtin_amdgcn_wmma_f32_16x16x32_bf16(false, afrag[1][s], false, b,
                                                     (short)0, c1, false, false);
      }
#pragma unroll
      for (int r = 0; r < 8; ++r) {
        s_hid[wave][khalf * 8 + r][n]      = (__bf16)fmaxf(c0[r], 0.0f);
        s_hid[wave][16 + khalf * 8 + r][n] = (__bf16)fmaxf(c1[r], 0.0f);
      }
    }
    __syncthreads();

    // ---- GEMM2: out = hidden @ w2 + b2 (G=10 padded to 16) ----
    {
      // w2T B fragments loaded once, reused for both row tiles.
      v16bf bw2[4];
#pragma unroll
      for (int s = 0; s < 4; ++s)
        bw2[s] = load_bfrag(w2T, col, s * 32 + khalf * 8);

      const float bias2 = (col < G_DIM) ? b2s[h][col] : 0.0f;
#pragma unroll
      for (int ra = 0; ra < 2; ++ra) {
        v8f c;
#pragma unroll
        for (int r = 0; r < 8; ++r) c[r] = bias2;
#pragma unroll
        for (int s = 0; s < 4; ++s) {
          const __bf16* pr = &s_hid[wave][ra * 16 + col][s * 32 + khalf * 8];
          v8bf lo = *(const v8bf*)pr;
          v8bf hi = *(const v8bf*)(pr + 16);
          v16bf a2 = __builtin_shufflevector(lo, hi, 0,1,2,3,4,5,6,7,8,9,10,11,12,13,14,15);
          c = __builtin_amdgcn_wmma_f32_16x16x32_bf16(false, a2, false, bw2[s],
                                                      (short)0, c, false, false);
        }
#pragma unroll
        for (int r = 0; r < 8; ++r)
          s_epi[wave][h][ra * 16 + khalf * 8 + r][col] = c[r];
      }
    }
    __syncthreads();
  }

  // ---- Per-row MDN epilogue: every lane owns one of the wave's 32 rows ----
  float p0, p1, p2;  // loss-num, loss-den, score partials
  {
    const int row = rowBase + lane;
    const float d = dist[row];
    const bool mk = (dist_mask[row] != 0) && (d <= 8.0f);
    const float mkf = mk ? 1.0f : 0.0f;

    float lp[G_DIM];
    float maxl = -INFINITY;
#pragma unroll
    for (int g = 0; g < G_DIM; ++g) {
      lp[g] = s_epi[wave][0][lane][g];
      maxl = fmaxf(maxl, lp[g]);
    }
    float se = 0.0f;
#pragma unroll
    for (int g = 0; g < G_DIM; ++g) { lp[g] = __expf(lp[g] - maxl); se += lp[g]; }
    const float inv = 1.0f / se;

    float t[G_DIM];
    float maxt = -INFINITY;
    float probsum = 0.0f;
    const size_t obase = (size_t)row * G_DIM;
#pragma unroll
    for (int g = 0; g < G_DIM; ++g) {
      float sraw = s_epi[wave][1][lane][g];
      float mraw = s_epi[wave][2][lane][g];
      float sig = (sraw > 0.0f ? sraw : __expf(sraw) - 1.0f) + 1.4f;  // elu+1.4
      float mu  = (mraw > 0.0f ? mraw : __expf(mraw) - 1.0f) + 1.0f;  // elu+1.0
      float z = (d - mu) / sig;
      float ll = -0.5f * z * z - __logf(sig) - 0.91893853320467274f;  // -0.5*log(2pi)
      float pig = lp[g] * inv;
      float tg = __logf(pig + 1e-10f) + ll;
      t[g] = tg;
      maxt = fmaxf(maxt, tg);
      probsum += __expf(ll) * pig;
      out_pi[obase + g]  = pig;
      out_mu[obase + g]  = mu;
      out_sig[obase + g] = sig;
    }
    float st = 0.0f;
#pragma unroll
    for (int g = 0; g < G_DIM; ++g) st += __expf(t[g] - maxt);
    const float closs = -(maxt + __logf(st));
    p0 = closs * mkf;
    p1 = mkf;
    p2 = (probsum / (d * d + 1e-10f)) * mkf;
  }
  // full-wave reduction
#pragma unroll
  for (int off = 16; off > 0; off >>= 1) {
    p0 += __shfl_down(p0, off, 32);
    p1 += __shfl_down(p1, off, 32);
    p2 += __shfl_down(p2, off, 32);
  }
  if (lane == 0) { s_red[wave][0] = p0; s_red[wave][1] = p1; s_red[wave][2] = p2; }
  __syncthreads();
  if (tid == 0) {
    float a0 = 0.0f, a1 = 0.0f, a2 = 0.0f;
    for (int w = 0; w < WAVES_PB; ++w) { a0 += s_red[w][0]; a1 += s_red[w][1]; a2 += s_red[w][2]; }
    atomicAdd(&lossacc[0], a0);
    atomicAdd(&lossacc[1], a1);
    const int bb = (blockIdx.x * BLOCK_ROWS) / NPTS;  // block spans a single batch index
    atomicAdd(&out_score[bb], a2);
  }
}

extern "C" void kernel_launch(void* const* d_in, const int* in_sizes, int n_in,
                              void* d_out, int out_size, void* d_ws, size_t ws_size,
                              hipStream_t stream) {
  const float* features          = (const float*)d_in[0];
  const float* dist              = (const float*)d_in[1];
  const unsigned char* dist_mask = (const unsigned char*)d_in[2];
  const float* w1_pi  = (const float*)d_in[3];
  const float* b1_pi  = (const float*)d_in[4];
  const float* w2_pi  = (const float*)d_in[5];
  const float* b2_pi  = (const float*)d_in[6];
  const float* w1_sig = (const float*)d_in[7];
  const float* b1_sig = (const float*)d_in[8];
  const float* w2_sig = (const float*)d_in[9];
  const float* b2_sig = (const float*)d_in[10];
  const float* w1_mu  = (const float*)d_in[11];
  const float* b1_mu  = (const float*)d_in[12];
  const float* w2_mu  = (const float*)d_in[13];
  const float* b2_mu  = (const float*)d_in[14];

  float* out = (float*)d_out;
  float* out_score = out;                                   // [32]
  float* out_loss  = out + NBATCH;                          // [1]
  float* out_pi    = out + NBATCH + 1;                      // [B,N,G]
  float* out_mu    = out_pi + (size_t)ROWS_TOTAL * G_DIM;   // [B,N,G]
  float* out_sig   = out_mu + (size_t)ROWS_TOTAL * G_DIM;   // [B,N,G]

  __bf16* ws_bf  = (__bf16*)d_ws;
  float* lossacc = (float*)((char*)d_ws + LOSSACC_BYTE_OFF);

  init_acc<<<1, 64, 0, stream>>>(out_score, lossacc);
  convert_weights<<<(WS_BF16_TOTAL + 255) / 256, 256, 0, stream>>>(
      w1_pi, w1_sig, w1_mu, w2_pi, w2_sig, w2_mu, ws_bf);
  mdn_fused<<<ROWS_TOTAL / BLOCK_ROWS, THREADS_PB, 0, stream>>>(
      features, dist, dist_mask, b1_pi, b2_pi, b1_sig, b2_sig, b1_mu, b2_mu,
      ws_bf, out_score, out_pi, out_mu, out_sig, lossacc);
  finalize_loss<<<1, 1, 0, stream>>>(lossacc, out_loss);
}
